// Encoder_86698209837455
// MI455X (gfx1250) — 16-bit and 8-bit matrix operand order repaired and checked by reading, compiled, not run on hardware
//
#include <hip/hip_runtime.h>
#include <hip/hip_bf16.h>

constexpr int kB = 4;
constexpr int kS = 2048;
constexpr int kE = 1024;
constexpr int kH = 16;
constexpr int kF = 4096;
constexpr int kD = 64;
constexpr int kM = kB * kS;

constexpr float kSX   = 64.0f;
constexpr float kSW   = 64.0f;
constexpr float kSCTX = 64.0f;
constexpr float kSH   = 16.0f;
constexpr float kSF1  = 64.0f;

typedef __attribute__((ext_vector_type(16))) _Float16     v16h;
typedef __attribute__((ext_vector_type(16))) int          v16i;
typedef __attribute__((ext_vector_type(8)))  float        v8f;
typedef __attribute__((ext_vector_type(4)))  unsigned int v4u;
typedef __attribute__((ext_vector_type(8)))  int          v8i;
typedef __attribute__((ext_vector_type(4)))  int          v4i;

union FragH { v16h v; uint4 q[2]; };
union Frag8 { v16i v; uint2 d[8]; uint4 q[4]; };

#ifdef __has_builtin
#if __has_builtin(__builtin_amdgcn_cvt_pk_fp8_f32)
#define USE_HW_FP8_CVT 1
#endif
#if __has_builtin(__builtin_amdgcn_tensor_load_to_lds)
#define USE_TDM 1
#endif
#endif

static __device__ __forceinline__ v8f wmma_f16(v16h a, v16h b, v8f c) {
  return __builtin_amdgcn_wmma_f32_16x16x32_f16(false, a, false, b,
                                                (short)0, c, false, false);
}
static __device__ __forceinline__ v8f wmma_fp8(v16i a, v16i b, v8f c) {
  return __builtin_amdgcn_wmma_f32_16x16x128_fp8_fp8(a, b, (short)0, c,
                                                     false, false);
}

static __device__ __forceinline__ v8f v8f_zero() {
  v8f z;
#pragma unroll
  for (int i = 0; i < 8; ++i) z[i] = 0.0f;
  return z;
}

static __device__ __forceinline__ unsigned char cvt_e4m3(float x) {
#if defined(USE_HW_FP8_CVT)
  return (unsigned char)(__builtin_amdgcn_cvt_pk_fp8_f32(x, x, 0, false) & 0xFF);
#else
  unsigned bits = __float_as_uint(x);
  unsigned s = (bits >> 31) << 7;
  float a = __uint_as_float(bits & 0x7FFFFFFFu);
  if (a != a) return (unsigned char)(s | 0x7F);
  if (a >= 464.0f) return (unsigned char)(s | 0x7E);
  if (a < 0.015625f) {
    int q = (int)(a * 512.0f + 0.5f);
    return (unsigned char)(s | (unsigned)q);
  }
  unsigned ab = bits & 0x7FFFFFFFu;
  int e = (int)(ab >> 23) - 126;
  float m = __uint_as_float((ab & 0x7FFFFFu) | 0x3F000000u);
  int q = (int)(m * 16.0f + 0.5f);
  if (q == 16) { q = 8; ++e; }
  int be = e + 6;
  if (be >= 16) return (unsigned char)(s | 0x7E);
  return (unsigned char)(s | (unsigned)(be << 3) | (unsigned)(q & 7));
#endif
}

__global__ __launch_bounds__(256)
void wt_fp8_kernel(const float* __restrict__ W, unsigned char* __restrict__ Wt,
                   int K, int N, float scale) {
  __shared__ float tile[32][33];
  int k0 = blockIdx.x * 32, n0 = blockIdx.y * 32;
  int tx = threadIdx.x & 31, ty = threadIdx.x >> 5;
#pragma unroll
  for (int i = ty; i < 32; i += 8)
    tile[i][tx] = W[(size_t)(k0 + i) * N + n0 + tx];
  __syncthreads();
#pragma unroll
  for (int i = ty; i < 32; i += 8)
    Wt[(size_t)(n0 + i) * K + k0 + tx] = cvt_e4m3(tile[tx][i] * scale);
}

__global__ void embed_kernel(const int* __restrict__ ids,
                             const float* __restrict__ emb,
                             float* __restrict__ xf,
                             unsigned char* __restrict__ xq) {
  int m = blockIdx.x;
  size_t row = (size_t)ids[m] * kE;
  size_t base = (size_t)m * kE;
  for (int i = threadIdx.x; i < kE; i += blockDim.x) {
    float v = emb[row + i];
    xf[base + i] = v;
    xq[base + i] = cvt_e4m3(v * kSX);
  }
}

#if defined(USE_TDM)
static __device__ __forceinline__ v8i tdm_group1(unsigned K) {
  v8i g;
  unsigned td1 = 1u << 20;
  g[0] = (int)((1u << 20) | (4u << 22) | (3u << 25));
  g[1] = (int)(K << 16);
  g[2] = (int)((K >> 16) | (td1 << 16));
  g[3] = (int)((td1 >> 16) | (128u << 16));
  g[4] = (int)128u;
  g[5] = (int)K;
  g[6] = 0;
  g[7] = 0;
  return g;
}
static __device__ __forceinline__ v4u tdm_group0(unsigned lds_off,
                                                 unsigned long long gaddr) {
  v4u g;
  g[0] = 1u;
  g[1] = lds_off;
  g[2] = (unsigned)gaddr;
  g[3] = (unsigned)(gaddr >> 32) | 0x80000000u;
  return g;
}
#endif

template <int ACT, int LAYOUT>
__global__ __launch_bounds__(256)
void gemm_fp8_kernel(const unsigned char* __restrict__ A,
                     const unsigned char* __restrict__ Bt,
                     const float* __restrict__ bias,
                     float* __restrict__ Cf, _Float16* __restrict__ Ch,
                     unsigned char* __restrict__ C8,
                     float invAB, float oscale, int N, int K) {
  __shared__ __align__(16) unsigned char As[128][144];
  __shared__ __align__(16) unsigned char Bs[128][144];

  const int tid  = threadIdx.x;
  const int m0   = blockIdx.y * 128;
  const int n0   = blockIdx.x * 128;
  const int lane = tid & 31;
  const int w    = tid >> 5;
  const int wm   = (w & 1) * 64;
  const int wn   = (w >> 1) * 32;
  const int lm   = lane & 15;
  const int kb8  = (lane < 16) ? 0 : 8;
  const int kb16 = (lane < 16) ? 0 : 16;

  v8f acc[4][2];
#pragma unroll
  for (int i = 0; i < 4; ++i)
#pragma unroll
    for (int j = 0; j < 2; ++j) acc[i][j] = v8f_zero();

#if defined(USE_TDM)
  const unsigned ldsA = (unsigned)(size_t)(void*)&As[0][0];
  const unsigned ldsB = (unsigned)(size_t)(void*)&Bs[0][0];
  const unsigned long long gA =
      (unsigned long long)(size_t)(A + (size_t)m0 * K);
  const unsigned long long gB =
      (unsigned long long)(size_t)(Bt + (size_t)n0 * K);
  const v8i g1 = tdm_group1((unsigned)K);
  const v4i z4 = {0, 0, 0, 0};
  const v8i z8 = {0, 0, 0, 0, 0, 0, 0, 0};
#else
  const int arow = tid >> 1;
  const int aseg = (tid & 1) * 64;
#endif

  for (int k0 = 0; k0 < K; k0 += 128) {
#if defined(USE_TDM)
    if (w == 0) {
      __builtin_amdgcn_tensor_load_to_lds(tdm_group0(ldsA, gA + (unsigned)k0),
                                          g1, z4, z4, z8, 0);
      __builtin_amdgcn_tensor_load_to_lds(tdm_group0(ldsB, gB + (unsigned)k0),
                                          g1, z4, z4, z8, 0);
      __builtin_amdgcn_s_wait_tensorcnt(0);
    }
#else
    {
      const uint4* src = (const uint4*)(A + (size_t)(m0 + arow) * K + k0 + aseg);
#pragma unroll
      for (int t = 0; t < 4; ++t)
        *(uint4*)&As[arow][aseg + 16 * t] = src[t];
      if (k0 + 128 < K)
        __builtin_prefetch(A + (size_t)(m0 + arow) * K + k0 + 128 + aseg, 0, 1);
    }
    {
      const uint4* src = (const uint4*)(Bt + (size_t)(n0 + arow) * K + k0 + aseg);
#pragma unroll
      for (int t = 0; t < 4; ++t)
        *(uint4*)&Bs[arow][aseg + 16 * t] = src[t];
      if (k0 + 128 < K)
        __builtin_prefetch(Bt + (size_t)(n0 + arow) * K + k0 + 128 + aseg, 0, 1);
    }
#endif
    __syncthreads();

    v16i bf[2];
#pragma unroll
    for (int j = 0; j < 2; ++j) {
      Frag8 u;
      int col = wn + 16 * j + lm;
#pragma unroll
      for (int t = 0; t < 4; ++t)
        u.q[t] = *(const uint4*)&Bs[col][kb16 + 32 * t];
      bf[j] = u.v;
    }
#pragma unroll
    for (int i = 0; i < 4; ++i) {
      Frag8 ua;
      int row = wm + 16 * i + lm;
#pragma unroll
      for (int sub = 0; sub < 2; ++sub)
#pragma unroll
        for (int t = 0; t < 4; ++t)
          ua.d[sub * 4 + t] = *(const uint2*)&As[row][64 * sub + kb16 + 32 * (t >> 1) + 8 * (t & 1)];
#pragma unroll
      for (int j = 0; j < 2; ++j)
        acc[i][j] = wmma_fp8(ua.v, bf[j], acc[i][j]);
    }
    __syncthreads();
  }

#pragma unroll
  for (int i = 0; i < 4; ++i) {
#pragma unroll
    for (int j = 0; j < 2; ++j) {
      int gn = n0 + wn + 16 * j + lm;
      float bv = bias ? bias[gn] : 0.0f;
#pragma unroll
      for (int r = 0; r < 8; ++r) {
        int gm = m0 + wm + 16 * i + r + ((lane < 16) ? 0 : 8);
        float v = acc[i][j][r] * invAB + bv;
        if (ACT == 1) v = 0.5f * v * (1.0f + erff(v * 0.70710678118654752f));
        if (LAYOUT == 0) {
          if (Cf) Cf[(size_t)gm * N + gn] = v;
          if (C8) C8[(size_t)gm * N + gn] = cvt_e4m3(v * oscale);
        } else {
          int bb = gm >> 11;
          int ss = gm & (kS - 1);
          int hh = gn >> 6;
          int dd = gn & (kD - 1);
          Ch[(((size_t)(bb * kH + hh)) * kS + ss) * kD + dd] = (_Float16)v;
        }
      }
    }
  }
}

__global__ __launch_bounds__(128)
void attn_kernel(const _Float16* __restrict__ Q,
                 const _Float16* __restrict__ Kh,
                 const _Float16* __restrict__ Vh,
                 const int* __restrict__ amask,
                 unsigned char* __restrict__ ctx8) {
  __shared__ __align__(16) _Float16 Qs[64][72];
  __shared__ __align__(16) _Float16 Ks[32][72];
  __shared__ __align__(16) _Float16 Vs[64][40];
  __shared__ __align__(16) _Float16 Ps[4][16][40];
  __shared__ int Ms[32];

  const int tid  = threadIdx.x;
  const int qt   = blockIdx.x * 64;
  const int h    = blockIdx.y;
  const int b    = blockIdx.z;
  const size_t head = ((size_t)(b * kH + h)) * kS * kD;
  const int lane = tid & 31;
  const int w    = tid >> 5;
  const int lm   = lane & 15;
  const int kbA  = (lane < 16) ? 0 : 8;
  const int kbB  = (lane < 16) ? 0 : 16;

#pragma unroll
  for (int it = 0; it < 4; ++it) {
    int id  = tid + it * 128;
    int row = id >> 3;
    int seg = (id & 7) * 8;
    *(uint4*)&Qs[row][seg] =
        *(const uint4*)(Q + head + (size_t)(qt + row) * kD + seg);
  }

  v8f ctx[4];
#pragma unroll
  for (int j = 0; j < 4; ++j) ctx[j] = v8f_zero();
  float m_run[8], l_run[8];
#pragma unroll
  for (int r = 0; r < 8; ++r) { m_run[r] = -1e30f; l_run[r] = 0.0f; }

  for (int kc = 0; kc < kS; kc += 32) {
    __syncthreads();
#pragma unroll
    for (int it = 0; it < 2; ++it) {
      int id  = tid + it * 128;
      int row = id >> 3;
      int seg = (id & 7) * 8;
      *(uint4*)&Ks[row][seg] =
          *(const uint4*)(Kh + head + (size_t)(kc + row) * kD + seg);
      uint4 d = *(const uint4*)(Vh + head + (size_t)(kc + row) * kD + seg);
      const _Float16* hp = (const _Float16*)&d;
#pragma unroll
      for (int e = 0; e < 8; ++e) Vs[seg + e][row] = hp[e];
    }
    if (tid < 32) Ms[tid] = amask[(size_t)b * kS + kc + tid];
    __syncthreads();

    v8f s0 = v8f_zero(), s1 = v8f_zero();
#pragma unroll
    for (int dk = 0; dk < 64; dk += 32) {
      FragH ua, ub0, ub1;
      int qrow = w * 16 + lm;
      ua.q[0]  = *(const uint4*)&Qs[qrow][dk + kbA];
      ua.q[1]  = *(const uint4*)&Qs[qrow][dk + 16 + kbA];
      ub0.q[0] = *(const uint4*)&Ks[lm][dk + kbA];
      ub0.q[1] = *(const uint4*)&Ks[lm][dk + 16 + kbA];
      ub1.q[0] = *(const uint4*)&Ks[16 + lm][dk + kbA];
      ub1.q[1] = *(const uint4*)&Ks[16 + lm][dk + 16 + kbA];
      s0 = wmma_f16(ua.v, ub0.v, s0);
      s1 = wmma_f16(ua.v, ub1.v, s1);
    }

    const float mb0 = Ms[lm]      ? 0.0f : -1e30f;
    const float mb1 = Ms[16 + lm] ? 0.0f : -1e30f;

#pragma unroll
    for (int r = 0; r < 8; ++r) {
      float v0 = s0[r] * 0.125f + mb0;
      float v1 = s1[r] * 0.125f + mb1;
      float mx = fmaxf(v0, v1);
#pragma unroll
      for (int o = 8; o >= 1; o >>= 1) mx = fmaxf(mx, __shfl_xor(mx, o, 32));
      float mN = fmaxf(m_run[r], mx);
      float al = __expf(m_run[r] - mN);
      float p0 = __expf(v0 - mN);
      float p1 = __expf(v1 - mN);
      float ps = p0 + p1;
#pragma unroll
      for (int o = 8; o >= 1; o >>= 1) ps += __shfl_xor(ps, o, 32);
      l_run[r] = l_run[r] * al + ps;
      m_run[r] = mN;
      int prow = r + ((lane < 16) ? 0 : 8);
      Ps[w][prow][lm]      = (_Float16)p0;
      Ps[w][prow][16 + lm] = (_Float16)p1;
#pragma unroll
      for (int j = 0; j < 4; ++j) ctx[j][r] *= al;
    }
    asm volatile("s_wait_dscnt 0" ::: "memory");

    FragH up;
    up.q[0] = *(const uint4*)&Ps[w][lm][kbA];
    up.q[1] = *(const uint4*)&Ps[w][lm][16 + kbA];
#pragma unroll
    for (int j = 0; j < 4; ++j) {
      FragH uv;
      int dd = j * 16 + lm;
      uv.q[0] = *(const uint4*)&Vs[dd][kbA];
      uv.q[1] = *(const uint4*)&Vs[dd][16 + kbA];
      ctx[j] = wmma_f16(up.v, uv.v, ctx[j]);
    }
  }

#pragma unroll
  for (int r = 0; r < 8; ++r) {
    float l = l_run[r];
    float inv = (l > 0.0f) ? (1.0f / l) : 0.0f;
    int sidx = qt + w * 16 + r + ((lane < 16) ? 0 : 8);
    size_t obase = ((size_t)(b * kS + sidx)) * kE + h * kD;
#pragma unroll
    for (int j = 0; j < 4; ++j)
      ctx8[obase + j * 16 + lm] = cvt_e4m3(ctx[j][r] * inv * kSCTX);
  }
}

__global__ __launch_bounds__(256)
void ln_kernel(const float* __restrict__ a, const float* __restrict__ x,
               const float* __restrict__ g, const float* __restrict__ beta,
               unsigned char* __restrict__ outq) {
  int m = blockIdx.x;
  const size_t base = (size_t)m * kE;
  __shared__ float red0[8], red1[8];
  float s = 0.0f, s2 = 0.0f;
  for (int i = threadIdx.x; i < kE; i += blockDim.x) {
    float v = a[base + i] + x[base + i];
    s += v; s2 += v * v;
  }
#pragma unroll
  for (int o = 16; o >= 1; o >>= 1) {
    s  += __shfl_xor(s, o, 32);
    s2 += __shfl_xor(s2, o, 32);
  }
  int w = threadIdx.x >> 5, lane = threadIdx.x & 31;
  if (lane == 0) { red0[w] = s; red1[w] = s2; }
  __syncthreads();
  if (threadIdx.x == 0) {
    float t = 0.0f, t2 = 0.0f;
    for (int i = 0; i < 8; ++i) { t += red0[i]; t2 += red1[i]; }
    red0[0] = t; red1[0] = t2;
  }
  __syncthreads();
  float mu  = red0[0] * (1.0f / kE);
  float var = red1[0] * (1.0f / kE) - mu * mu;
  float inv = rsqrtf(var + 1e-5f);
  for (int i = threadIdx.x; i < kE; i += blockDim.x) {
    float v = a[base + i] + x[base + i];
    outq[base + i] = cvt_e4m3(((v - mu) * inv * g[i] + beta[i]) * kSH);
  }
}

__global__ __launch_bounds__(128)
void pool_dot_kernel(const float* __restrict__ ff, const float* __restrict__ Wp,
                     const float* __restrict__ bp, float* __restrict__ logits) {
  int m = blockIdx.x;
  const size_t base = (size_t)m * kE;
  float p0 = 0.0f, p1 = 0.0f, p2 = 0.0f;
  for (int i = threadIdx.x; i < kE; i += blockDim.x) {
    float v = ff[base + i];
    p0 += v * Wp[i * 3 + 0];
    p1 += v * Wp[i * 3 + 1];
    p2 += v * Wp[i * 3 + 2];
  }
#pragma unroll
  for (int o = 16; o >= 1; o >>= 1) {
    p0 += __shfl_xor(p0, o, 32);
    p1 += __shfl_xor(p1, o, 32);
    p2 += __shfl_xor(p2, o, 32);
  }
  __shared__ float red[4][3];
  int w = threadIdx.x >> 5, lane = threadIdx.x & 31;
  if (lane == 0) { red[w][0] = p0; red[w][1] = p1; red[w][2] = p2; }
  __syncthreads();
  if (threadIdx.x == 0) {
    float t0 = 0.0f, t1 = 0.0f, t2 = 0.0f;
    for (int i = 0; i < 4; ++i) { t0 += red[i][0]; t1 += red[i][1]; t2 += red[i][2]; }
    logits[(size_t)m * 3 + 0] = t0 + bp[0];
    logits[(size_t)m * 3 + 1] = t1 + bp[1];
    logits[(size_t)m * 3 + 2] = t2 + bp[2];
  }
}

__global__ __launch_bounds__(256)
void pool_reduce_kernel(const float* __restrict__ logits,
                        float* __restrict__ out) {
  int b = blockIdx.x / 3, j = blockIdx.x % 3;
  float s = 0.0f;
  for (int si = threadIdx.x; si < kS; si += blockDim.x)
    s += logits[((size_t)(b * kS + si)) * 3 + j];
#pragma unroll
  for (int o = 16; o >= 1; o >>= 1) s += __shfl_xor(s, o, 32);
  __shared__ float red[8];
  int w = threadIdx.x >> 5, lane = threadIdx.x & 31;
  if (lane == 0) red[w] = s;
  __syncthreads();
  if (threadIdx.x == 0) {
    float t = 0.0f;
    for (int i = 0; i < 8; ++i) t += red[i];
    out[blockIdx.x] = t * (1.0f / kS);
  }
}

extern "C" void kernel_launch(void* const* d_in, const int* in_sizes, int n_in,
                              void* d_out, int out_size, void* d_ws, size_t ws_size,
                              hipStream_t stream) {
  (void)in_sizes; (void)n_in; (void)out_size; (void)ws_size;

  const int*   ids   = (const int*)d_in[0];
  const int*   amask = (const int*)d_in[1];
  const float* emb   = (const float*)d_in[2];
  const float* Wq    = (const float*)d_in[3];
  const float* bq    = (const float*)d_in[4];
  const float* Wk    = (const float*)d_in[5];
  const float* bk    = (const float*)d_in[6];
  const float* Wv    = (const float*)d_in[7];
  const float* bv    = (const float*)d_in[8];
  const float* Wo    = (const float*)d_in[9];
  const float* bo    = (const float*)d_in[10];
  const float* ln_g  = (const float*)d_in[11];
  const float* ln_b  = (const float*)d_in[12];
  const float* W1    = (const float*)d_in[13];
  const float* b1    = (const float*)d_in[14];
  const float* W2    = (const float*)d_in[15];
  const float* b2    = (const float*)d_in[16];
  const float* Wp    = (const float*)d_in[17];
  const float* bp    = (const float*)d_in[18];

  char* ws = (char*)d_ws;
  size_t off = 0;
  auto alloc = [&](size_t bytes) -> void* {
    void* p = ws + off;
    off = (off + bytes + 255) & ~(size_t)255;
    return p;
  };

  float*         xf    = (float*)        alloc((size_t)kM * kE * 4);
  unsigned char* xq    = (unsigned char*)alloc((size_t)kM * kE);
  unsigned char* wq8   = (unsigned char*)alloc((size_t)kE * kE);
  unsigned char* wk8   = (unsigned char*)alloc((size_t)kE * kE);
  unsigned char* wv8   = (unsigned char*)alloc((size_t)kE * kE);
  unsigned char* wo8   = (unsigned char*)alloc((size_t)kE * kE);
  unsigned char* w18   = (unsigned char*)alloc((size_t)kE * kF);
  unsigned char* w28   = (unsigned char*)alloc((size_t)kF * kE);
  _Float16*      qh    = (_Float16*)     alloc((size_t)kM * kE * 2);
  _Float16*      kh    = (_Float16*)     alloc((size_t)kM * kE * 2);
  _Float16*      vh    = (_Float16*)     alloc((size_t)kM * kE * 2);
  unsigned char* ctx8  = (unsigned char*)alloc((size_t)kM * kE);
  float*         ao    = (float*)        alloc((size_t)kM * kE * 4);
  unsigned char* h8    = (unsigned char*)alloc((size_t)kM * kE);
  unsigned char* f1q   = (unsigned char*)alloc((size_t)kM * kF);
  float*         f2f   = (float*)        alloc((size_t)kM * kE * 4);
  float*         lg    = (float*)        alloc((size_t)kM * 3 * 4);

  wt_fp8_kernel<<<dim3(kE / 32, kE / 32), 256, 0, stream>>>(Wq, wq8, kE, kE, kSW);
  wt_fp8_kernel<<<dim3(kE / 32, kE / 32), 256, 0, stream>>>(Wk, wk8, kE, kE, kSW);
  wt_fp8_kernel<<<dim3(kE / 32, kE / 32), 256, 0, stream>>>(Wv, wv8, kE, kE, kSW);
  wt_fp8_kernel<<<dim3(kE / 32, kE / 32), 256, 0, stream>>>(Wo, wo8, kE, kE, kSW);
  wt_fp8_kernel<<<dim3(kE / 32, kF / 32), 256, 0, stream>>>(W1, w18, kE, kF, kSW);
  wt_fp8_kernel<<<dim3(kF / 32, kE / 32), 256, 0, stream>>>(W2, w28, kF, kE, kSW);

  embed_kernel<<<kM, 256, 0, stream>>>(ids, emb, xf, xq);

  dim3 gEE(kE / 128, kM / 128);
  const float invXW = 1.0f / (kSX * kSW);
  gemm_fp8_kernel<0, 1><<<gEE, 256, 0, stream>>>(xq, wq8, bq, nullptr, qh,
                                                 nullptr, invXW, 1.0f, kE, kE);
  gemm_fp8_kernel<0, 1><<<gEE, 256, 0, stream>>>(xq, wk8, bk, nullptr, kh,
                                                 nullptr, invXW, 1.0f, kE, kE);
  gemm_fp8_kernel<0, 1><<<gEE, 256, 0, stream>>>(xq, wv8, bv, nullptr, vh,
                                                 nullptr, invXW, 1.0f, kE, kE);

  attn_kernel<<<dim3(kS / 64, kH, kB), 128, 0, stream>>>(qh, kh, vh, amask, ctx8);

  gemm_fp8_kernel<0, 0><<<gEE, 256, 0, stream>>>(
      ctx8, wo8, bo, ao, nullptr, nullptr, 1.0f / (kSCTX * kSW), 1.0f, kE, kE);

  ln_kernel<<<kM, 256, 0, stream>>>(ao, xf, ln_g, ln_b, h8);

  gemm_fp8_kernel<1, 0><<<dim3(kF / 128, kM / 128), 256, 0, stream>>>(
      h8, w18, b1, nullptr, nullptr, f1q, 1.0f / (kSH * kSW), kSF1, kF, kE);

  gemm_fp8_kernel<0, 0><<<gEE, 256, 0, stream>>>(
      f1q, w28, b2, f2f, nullptr, nullptr, 1.0f / (kSF1 * kSW), 1.0f, kE, kF);

  pool_dot_kernel<<<kM, 128, 0, stream>>>(f2f, Wp, bp, lg);
  pool_reduce_kernel<<<kB * 3, 256, 0, stream>>>(lg, (float*)d_out);
}
